// GladiusAttention_41326175322675
// MI455X (gfx1250) — matrix operand order repaired and checked by reading, not compiled, not run on hardware
//
#include <hip/hip_runtime.h>
#include <hip/hip_bf16.h>
#include <math.h>

// ---------------------------------------------------------------------------
// CDNA5 / gfx1250 attention layer: WMMA everywhere + TDM (tensor_load_to_lds)
// ---------------------------------------------------------------------------

typedef __attribute__((ext_vector_type(16))) _Float16 v16h;
typedef __attribute__((ext_vector_type(8)))  _Float16 v8h;
typedef __attribute__((ext_vector_type(8)))  float    v8f;
typedef __attribute__((ext_vector_type(4)))  unsigned int u32x4;
typedef __attribute__((ext_vector_type(8)))  int      i32x8;
typedef __attribute__((ext_vector_type(4)))  int      i32x4;

union V16U { v16h v; v8h h[2]; };

static __device__ __forceinline__ v8f wmma16(v16h a, v16h b, v8f c) {
    // D = A(16x32 f16) * B(32x16 f16) + C(16x16 f32)
    return __builtin_amdgcn_wmma_f32_16x16x32_f16(false, a, false, b,
                                                  (short)0, c, false, false);
}

// TDM issue: D# group0 (4 dw) + group1 (8 dw); groups 2/3 disabled (2D tile).
static __device__ __forceinline__ void tdm_load(u32x4 g0, i32x8 g1) {
    i32x4 z4 = {0, 0, 0, 0};
#if defined(__clang_major__) && (__clang_major__ >= 23)
    i32x8 z8 = {0, 0, 0, 0, 0, 0, 0, 0};
    __builtin_amdgcn_tensor_load_to_lds(g0, g1, z4, z4, z8, 0);
#else
    __builtin_amdgcn_tensor_load_to_lds(g0, g1, z4, z4, 0);
#endif
}

#define BATCH 4
#define SEQ   2048
#define HID   1024
#define NH    16
#define NKV   4
#define HD    64
#define GROUPS (NH / NKV)

// ---------------------------------------------------------------------------
// elementwise conversion / layout kernels
// ---------------------------------------------------------------------------
__global__ void k_f32_to_f16(const float* __restrict__ src,
                             _Float16* __restrict__ dst, int n) {
    int i = blockIdx.x * blockDim.x + threadIdx.x;
    if (i < n) dst[i] = (_Float16)src[i];
}

// wt[n*K + k] = w[k*N + n]  (transpose + f16 convert)
__global__ void k_convert_transpose(const float* __restrict__ w,
                                    _Float16* __restrict__ wt, int K, int N) {
    int i = blockIdx.x * blockDim.x + threadIdx.x;
    if (i >= K * N) return;
    int n = i % N, k = i / N;
    wt[(size_t)n * K + k] = (_Float16)w[i];
}

// RoPE + head-major rearrange: src [B*S, NHD*64] -> dst [B, NHD, S, 64]
__global__ void k_rope(const _Float16* __restrict__ src,
                       _Float16* __restrict__ dst, int n_heads) {
    int idx = blockIdx.x * blockDim.x + threadIdx.x;
    int total = BATCH * SEQ * n_heads * 32;
    if (idx >= total) return;
    int i = idx & 31; int t = idx >> 5;
    int h = t % n_heads; t /= n_heads;
    int s = t % SEQ;     int b = t / SEQ;

    float f   = __powf(10000.0f, -(float)i / 32.0f);
    float ang = (float)s * f;
    float c = cosf(ang), sn = sinf(ang);

    size_t src_row = ((size_t)b * SEQ + s) * ((size_t)n_heads * HD) + (size_t)h * HD;
    float lo = (float)src[src_row + i];
    float hi = (float)src[src_row + i + 32];

    size_t dst_row = (((size_t)b * n_heads + h) * SEQ + s) * HD;
    dst[dst_row + i]      = (_Float16)(lo * c - hi * sn);
    dst[dst_row + i + 32] = (_Float16)(hi * c + lo * sn);
}

// V transpose: src [B*S, NKV*64] -> dst [B, NKV, 64, S]
__global__ void k_vtrans(const _Float16* __restrict__ src,
                         _Float16* __restrict__ dst) {
    int idx = blockIdx.x * blockDim.x + threadIdx.x;
    int total = BATCH * NKV * HD * SEQ;
    if (idx >= total) return;
    int s = idx % SEQ; int t = idx / SEQ;
    int d = t % HD;    t /= HD;
    int kh = t % NKV;  int b = t / NKV;
    dst[idx] = src[((size_t)b * SEQ + s) * (NKV * HD) + (size_t)kh * HD + d];
}

// ---------------------------------------------------------------------------
// WMMA GEMM: C[M,N] = A[M,K] * Bt[N,K]^T
// Block = 8 waves, 256M x 64N per block (32M x 64N per wave).
// B tile (64 rows x 32 K, 4KB) fetched ONCE per block per k-step by the
// Tensor Data Mover into double-buffered LDS; A double-buffered in registers.
// ---------------------------------------------------------------------------
__global__ void k_gemm_wmma(const _Float16* __restrict__ A,
                            const _Float16* __restrict__ Bt,
                            void* __restrict__ Cout,
                            int M, int N, int K, int out_f32) {
    __shared__ _Float16 lds_b[2][64 * 32];               // 2 x 4KB
    const int wave = threadIdx.x >> 5;
    const int lane = threadIdx.x & 31;
    const int l16  = lane & 15;
    const int half = lane >> 4;
    const int m0 = blockIdx.x * 256 + wave * 32;
    const int n0 = blockIdx.y * 64;
    const int nk = K / 32;

    v8f acc[2][4];
#pragma unroll
    for (int a = 0; a < 2; ++a)
#pragma unroll
        for (int nt = 0; nt < 4; ++nt) acc[a][nt] = (v8f)0.0f;

    const _Float16* arow0 = A + (size_t)(m0 + l16) * K;
    const _Float16* arow1 = A + (size_t)(m0 + 16 + l16) * K;

    // --- TDM descriptor builder: 2D tile (tile_dim0=32 K-elems, tile_dim1=64 rows)
    auto tdm_issue = [&](int kb, int buf) {
        unsigned long long gaddr =
            (unsigned long long)(uintptr_t)(Bt + (size_t)n0 * K + (size_t)kb * 32);
        unsigned int lds_off = (unsigned int)(uintptr_t)(&lds_b[buf][0]);
        u32x4 g0;
        g0[0] = 1u;                                       // count=1, user mode
        g0[1] = lds_off;                                  // lds_addr (bytes)
        g0[2] = (unsigned int)gaddr;                      // global_addr[31:0]
        g0[3] = (unsigned int)((gaddr >> 32) & 0x1FFFFFFu) | (2u << 30); // type=2
        unsigned int Ku = (unsigned int)K, Nu = (unsigned int)N;
        i32x8 g1;
        g1[0] = (int)(1u << 16);                          // data_size=1 (2 bytes)
        g1[1] = (int)(Ku << 16);                          // tensor_dim0[15:0]
        g1[2] = (int)((Ku >> 16) | (Nu << 16));           // tdim0[31:16] | tdim1[15:0]
        g1[3] = (int)((Nu >> 16) | (32u << 16));          // tdim1[31:16] | tile_dim0=32
        g1[4] = (int)64u;                                 // tile_dim1=64, tile_dim2=0
        g1[5] = (int)Ku;                                  // tensor_dim0_stride[31:0]
        g1[6] = 0;                                        // stride0 hi | stride1 lo
        g1[7] = 0;                                        // stride1 hi
        tdm_load(g0, g1);
    };

    if (wave == 0) tdm_issue(0, 0);                       // prologue DMA

    V16U a_cur[2], a_nxt[2];
    a_cur[0].v = *(const v16h*)(arow0 + half * 16);
    a_cur[1].v = *(const v16h*)(arow1 + half * 16);

    for (int kb = 0; kb < nk; ++kb) {
        if (wave == 0) __builtin_amdgcn_s_wait_tensorcnt(0);   // tile kb landed
        __syncthreads();
        if (wave == 0 && kb + 1 < nk) tdm_issue(kb + 1, (kb + 1) & 1);

        a_nxt[0] = a_cur[0]; a_nxt[1] = a_cur[1];
        if (kb + 1 < nk) {                                 // uniform branch
            int k = (kb + 1) * 32;
            a_nxt[0].v = *(const v16h*)(arow0 + k + half * 16);
            a_nxt[1].v = *(const v16h*)(arow1 + k + half * 16);
        }

        const _Float16* bb = &lds_b[kb & 1][0];
        v16h b[4];
#pragma unroll
        for (int nt = 0; nt < 4; ++nt)
            b[nt] = *(const v16h*)(bb + (nt * 16 + l16) * 32 + half * 16);
#pragma unroll
        for (int nt = 0; nt < 4; ++nt) {
            acc[0][nt] = wmma16(a_cur[0].v, b[nt], acc[0][nt]);
            acc[1][nt] = wmma16(a_cur[1].v, b[nt], acc[1][nt]);
        }
        __syncthreads();                                   // tile kb free
        a_cur[0] = a_nxt[0]; a_cur[1] = a_nxt[1];
    }

#pragma unroll
    for (int a = 0; a < 2; ++a)
#pragma unroll
        for (int nt = 0; nt < 4; ++nt)
#pragma unroll
            for (int r = 0; r < 8; ++r) {
                size_t row = (size_t)(m0 + a * 16 + r + 8 * half);
                size_t col = (size_t)(n0 + nt * 16 + l16);
                if (out_f32) ((float*)Cout)[row * N + col] = acc[a][nt][r];
                else ((_Float16*)Cout)[row * N + col] = (_Float16)acc[a][nt][r];
            }
}

// ---------------------------------------------------------------------------
// Flash attention: one wave per 16-row q tile; 32-key blocks; online softmax
// qbuf [B,NH,S,64]   kbuf [B,NKV,S,64]   vT [B,NKV,64,S]   obuf [B,S,NH*64]
// ---------------------------------------------------------------------------
__global__ void k_flash(const _Float16* __restrict__ qbuf,
                        const _Float16* __restrict__ kbuf,
                        const _Float16* __restrict__ vT,
                        _Float16* __restrict__ obuf) {
    __shared__ _Float16 lds_p[8][16][32];                // per-wave P staging
    const int wave = threadIdx.x >> 5;
    const int lane = threadIdx.x & 31;
    const int l16  = lane & 15;
    const int half = lane >> 4;

    const int tile = blockIdx.x * 8 + wave;
    const int t0 = tile * 16;
    const int h = blockIdx.y;
    const int b = blockIdx.z;
    const int kh = h / GROUPS;

    const _Float16* qb = qbuf + ((size_t)(b * NH + h)) * SEQ * HD;
    const _Float16* kb = kbuf + ((size_t)(b * NKV + kh)) * SEQ * HD;
    const _Float16* vb = vT   + ((size_t)(b * NKV + kh)) * HD * SEQ;

    const _Float16* qrow = qb + (size_t)(t0 + l16) * HD;
    V16U aq0, aq1;
    aq0.v = *(const v16h*)(qrow + half * 16);
    aq1.v = *(const v16h*)(qrow + 32 + half * 16);

    float m_i[8], l_i[8];
    v8f o_acc[4];
#pragma unroll
    for (int r = 0; r < 8; ++r) { m_i[r] = -1e30f; l_i[r] = 0.0f; }
#pragma unroll
    for (int nd = 0; nd < 4; ++nd) o_acc[nd] = (v8f)0.0f;

    const int nkb = (t0 + 47) >> 5;                      // ceil((t0+16)/32)
    for (int kbk = 0; kbk < nkb; ++kbk) {
        const int k0 = kbk * 32;
        __builtin_prefetch(kb + (size_t)(k0 + 32 + l16) * HD, 0, 3);

        v8f sc[2];
#pragma unroll
        for (int nt = 0; nt < 2; ++nt) {
            sc[nt] = (v8f)0.0f;
            const _Float16* krow = kb + (size_t)(k0 + nt * 16 + l16) * HD;
            v16h bk0 = *(const v16h*)(krow + half * 16);
            v16h bk1 = *(const v16h*)(krow + 32 + half * 16);
            sc[nt] = wmma16(aq0.v, bk0, sc[nt]);
            sc[nt] = wmma16(aq1.v, bk1, sc[nt]);
        }

        const bool need_mask = (k0 + 31) > t0;           // uniform branch
#pragma unroll
        for (int nt = 0; nt < 2; ++nt)
#pragma unroll
            for (int r = 0; r < 8; ++r) {
                float v = sc[nt][r] * 0.125f;            // 1/sqrt(64)
                if (need_mask) {
                    int row = t0 + r + 8 * half;
                    int col = k0 + nt * 16 + l16;
                    if (col > row) v = -1e30f;
                }
                sc[nt][r] = v;
            }

#pragma unroll
        for (int r = 0; r < 8; ++r) {
            float mx = fmaxf(sc[0][r], sc[1][r]);
            mx = fmaxf(mx, __shfl_xor(mx, 1, 32));
            mx = fmaxf(mx, __shfl_xor(mx, 2, 32));
            mx = fmaxf(mx, __shfl_xor(mx, 4, 32));
            mx = fmaxf(mx, __shfl_xor(mx, 8, 32));
            float m_new = fmaxf(m_i[r], mx);
            float scl = __expf(m_i[r] - m_new);
            float p0 = __expf(sc[0][r] - m_new);
            float p1 = __expf(sc[1][r] - m_new);
            sc[0][r] = p0; sc[1][r] = p1;
            float rs = p0 + p1;
            rs += __shfl_xor(rs, 1, 32);
            rs += __shfl_xor(rs, 2, 32);
            rs += __shfl_xor(rs, 4, 32);
            rs += __shfl_xor(rs, 8, 32);
            l_i[r] = l_i[r] * scl + rs;
            m_i[r] = m_new;
#pragma unroll
            for (int nd = 0; nd < 4; ++nd) o_acc[nd][r] *= scl;
        }

        // C-layout -> A-layout via per-wave LDS
#pragma unroll
        for (int nt = 0; nt < 2; ++nt)
#pragma unroll
            for (int r = 0; r < 8; ++r)
                lds_p[wave][r + 8 * half][nt * 16 + l16] = (_Float16)sc[nt][r];
        asm volatile("s_wait_dscnt 0" ::: "memory");

        V16U ap;
        const _Float16* prow = &lds_p[wave][l16][0];
        ap.v = *(const v16h*)(prow + half * 16);

#pragma unroll
        for (int nd = 0; nd < 4; ++nd) {
            const _Float16* vrow =
                vb + (size_t)(nd * 16 + l16) * SEQ + k0 + half * 16;
            v16h bv = *(const v16h*)vrow;
            o_acc[nd] = wmma16(ap.v, bv, o_acc[nd]);
        }
    }

    _Float16* ob = obuf + (size_t)b * SEQ * (NH * HD) + (size_t)h * HD;
#pragma unroll
    for (int r = 0; r < 8; ++r) {
        float inv = 1.0f / l_i[r];
#pragma unroll
        for (int nd = 0; nd < 4; ++nd) {
            size_t row = (size_t)(t0 + r + 8 * half);
            ob[row * (NH * HD) + nd * 16 + l16] =
                (_Float16)(o_acc[nd][r] * inv);
        }
    }
}

// ---------------------------------------------------------------------------
// host launcher
// ---------------------------------------------------------------------------
extern "C" void kernel_launch(void* const* d_in, const int* in_sizes, int n_in,
                              void* d_out, int out_size, void* d_ws, size_t ws_size,
                              hipStream_t stream) {
    const float* x  = (const float*)d_in[0];
    const float* wq = (const float*)d_in[1];
    const float* wk = (const float*)d_in[2];
    const float* wv = (const float*)d_in[3];
    const float* wo = (const float*)d_in[4];
    float* out = (float*)d_out;

    const size_t M = (size_t)BATCH * SEQ;                // 8192 rows
    char* ws = (char*)d_ws;
    size_t off = 0;
    auto alloc = [&](size_t bytes) { char* p = ws + off; off += bytes; return p; };

    _Float16* xh   = (_Float16*)alloc(M * HID * 2);
    _Float16* qtmp = (_Float16*)alloc(M * (NH * HD) * 2);   // aliased as obuf later
    _Float16* ktmp = (_Float16*)alloc(M * (NKV * HD) * 2);
    _Float16* vtmp = (_Float16*)alloc(M * (NKV * HD) * 2);
    _Float16* qbuf = (_Float16*)alloc(M * (NH * HD) * 2);
    _Float16* kbuf = (_Float16*)alloc(M * (NKV * HD) * 2);
    _Float16* vbT  = (_Float16*)alloc(M * (NKV * HD) * 2);
    _Float16* wqT  = (_Float16*)alloc((size_t)HID * (NH * HD) * 2);
    _Float16* wkT  = (_Float16*)alloc((size_t)HID * (NKV * HD) * 2);
    _Float16* wvT  = (_Float16*)alloc((size_t)HID * (NKV * HD) * 2);
    _Float16* woT  = (_Float16*)alloc((size_t)(NH * HD) * HID * 2);
    _Float16* obuf = qtmp;                                 // reuse after RoPE
    (void)ws_size; (void)n_in; (void)in_sizes; (void)out_size;

    // 1) precision conversion + weight transposes
    {
        int n = (int)(M * HID);
        k_f32_to_f16<<<(n + 255) / 256, 256, 0, stream>>>(x, xh, n);
        int nq = HID * NH * HD;
        k_convert_transpose<<<(nq + 255) / 256, 256, 0, stream>>>(wq, wqT, HID, NH * HD);
        int nk = HID * NKV * HD;
        k_convert_transpose<<<(nk + 255) / 256, 256, 0, stream>>>(wk, wkT, HID, NKV * HD);
        k_convert_transpose<<<(nk + 255) / 256, 256, 0, stream>>>(wv, wvT, HID, NKV * HD);
        int no = NH * HD * HID;
        k_convert_transpose<<<(no + 255) / 256, 256, 0, stream>>>(wo, woT, NH * HD, HID);
    }

    // 2) QKV projections (f16 WMMA + TDM-staged weights, f16 out)
    k_gemm_wmma<<<dim3(M / 256, (NH * HD) / 64), 256, 0, stream>>>(
        xh, wqT, qtmp, (int)M, NH * HD, HID, 0);
    k_gemm_wmma<<<dim3(M / 256, (NKV * HD) / 64), 256, 0, stream>>>(
        xh, wkT, ktmp, (int)M, NKV * HD, HID, 0);
    k_gemm_wmma<<<dim3(M / 256, (NKV * HD) / 64), 256, 0, stream>>>(
        xh, wvT, vtmp, (int)M, NKV * HD, HID, 0);

    // 3) RoPE + layout for attention
    {
        int nq = BATCH * SEQ * NH * 32;
        k_rope<<<(nq + 255) / 256, 256, 0, stream>>>(qtmp, qbuf, NH);
        int nk = BATCH * SEQ * NKV * 32;
        k_rope<<<(nk + 255) / 256, 256, 0, stream>>>(ktmp, kbuf, NKV);
        int nv = BATCH * NKV * HD * SEQ;
        k_vtrans<<<(nv + 255) / 256, 256, 0, stream>>>(vtmp, vbT);
    }

    // 4) causal flash attention (WMMA for QK^T and PV)
    k_flash<<<dim3(SEQ / 16 / 8, NH, BATCH), 256, 0, stream>>>(qbuf, kbuf, vbT, obuf);

    // 5) output projection (f32 out -> d_out)
    k_gemm_wmma<<<dim3(M / 256, HID / 64), 256, 0, stream>>>(
        obuf, woT, out, (int)M, HID, NH * HD, 1);
}
